// DeepLSTM_74672301408704
// MI455X (gfx1250) — matrix operand order repaired and checked by reading, not compiled, not run on hardware
//
#include <hip/hip_runtime.h>

// ---------------------------------------------------------------------------
// DeepLSTM for MI455X (gfx1250, wave32, WMMA bf16 16x16x32 + TDM staging).
//
//   K0 conv_w : W_l (fp32, [K,1024]) -> Wt_l (bf16, N-major [1024][K])   (once)
//   K1 gather : Xbf[t*32+b][256] = bf16(emb[inputs[b][t]])               (parallel)
//   K2 phase1 : Zx[l][t*32+b][1024] = bf16(X @ W_l[0:256,:] + b_l)       (parallel WMMA)
//   K3 lstm   : single 1024-thread (32 wave32) persistent workgroup walks
//               t=0..1999; recurrent (32,256)@(256,1024) GEMMs via WMMA;
//               h state + gate-exchange buffer in LDS; per-stage Zx tile
//               (64KB) DMA'd into LDS by the Tensor Data Mover one stage
//               ahead (double buffered, TENSORcnt-tracked).
//   LDS: 48KB static (h) + 256KB dynamic (2x64KB Zx stage + 128KB z) = 304KB
//        of the 320KB/WGP CDNA5 allocation.
// ---------------------------------------------------------------------------

#define BATCHN 32
#define SZ     256
#define NG     1024
#define TSTEPS 2000
#define ROWS   (TSTEPS*BATCHN)   // 64000
#define NSTAGE (TSTEPS*3)        // 6000 sequential layer-stages

typedef __attribute__((ext_vector_type(16))) __bf16 v16bf;
typedef __attribute__((ext_vector_type(8)))  float  v8f;
typedef __attribute__((ext_vector_type(4)))  unsigned int u32x4;
typedef __attribute__((ext_vector_type(8)))  int  i32x8;
typedef __attribute__((ext_vector_type(4)))  int  i32x4;

#if defined(__HIP_DEVICE_COMPILE__) && \
    __has_builtin(__builtin_amdgcn_tensor_load_to_lds) && \
    __has_builtin(__builtin_amdgcn_s_wait_tensorcnt)
#define HAVE_TDM 1
#else
#define HAVE_TDM 0
#endif

union BF16Frag { v16bf v; int4 q[2]; };

__device__ __forceinline__ unsigned short f2bf(float f) {
    unsigned int u = __float_as_uint(f);
    unsigned int r = ((u >> 16) & 1u) + 0x7FFFu;   // round-to-nearest-even
    return (unsigned short)((u + r) >> 16);
}
__device__ __forceinline__ float bf2f(unsigned short h) {
    return __uint_as_float(((unsigned int)h) << 16);
}
__device__ __forceinline__ float sigm(float x) {
    return 1.0f / (1.0f + __expf(-x));
}
__device__ __forceinline__ float ftanh(float x) {
    x = fminf(15.0f, fmaxf(-15.0f, x));
    float e = __expf(2.0f * x);
    return (e - 1.0f) / (e + 1.0f);
}

__device__ __forceinline__ v8f wmma_bf16(v16bf a, v16bf b, v8f c) {
    return __builtin_amdgcn_wmma_f32_16x16x32_bf16(
        false, a, false, b, (short)0, c, false, false);
}

#if HAVE_TDM
// ---------------------------------------------------------------------------
// TDM: flat 1-D tile of 32768 bf16 elements (64KB) global -> LDS.
// D# per CDNA5 ISA ch.8: group0 {count=1 | lds_addr | global_addr | type=2},
// group1 {mask=0, data_size=2B, tensor_dim0=tile_dim0=32768, stride=32768},
// groups 2/3 zero (<=2-D tensor).  Issued per-wave; tracked with TENSORcnt.
// ---------------------------------------------------------------------------
__device__ __forceinline__ void tdm_load_tile(const unsigned short* gsrc,
                                              unsigned lds_byte_off) {
    unsigned long long ga = (unsigned long long)(const void*)gsrc;
    u32x4 g0 = { 1u,                                   // count=1, user mode
                 lds_byte_off,                         // D#.lds_addr
                 (unsigned)(ga & 0xFFFFFFFFu),         // global_addr[31:0]
                 (unsigned)((ga >> 32) & 0x1FFFFFFu)   // global_addr[56:32]
                   | (2u << 30) };                     // type=2 ("image")
    i32x8 g1 = { (int)(1u << 16),          // wg_mask=0, data_size=1 (2 bytes)
                 (int)(0x8000u << 16),     // tensor_dim0[15:0]=32768 in [31:16]
                 0,                        // tensor_dim0[31:16]=0, dim1 lo=0
                 (int)(0x8000u << 16),     // tile_dim0=32768 in [31:16]
                 0,                        // tile_dim1=0, tile_dim2=0 (1-D)
                 32768,                    // tensor_dim0_stride lo32
                 0, 0 };
    i32x4 z4 = { 0, 0, 0, 0 };
#if defined(__clang_major__) && (__clang_major__ >= 23)
    i32x8 z8 = { 0, 0, 0, 0, 0, 0, 0, 0 };
    __builtin_amdgcn_tensor_load_to_lds(g0, g1, z4, z4, z8, 0);
#else
    __builtin_amdgcn_tensor_load_to_lds(g0, g1, z4, z4, 0);
#endif
}
#endif // HAVE_TDM

// ---------------------------------------------------------------------------
// K0: transpose + convert weights to bf16, N-major: Wt[j][k] = bf16(W[k][j]).
// ---------------------------------------------------------------------------
__global__ void conv_w_kernel(const float* __restrict__ W0,
                              const float* __restrict__ W1,
                              const float* __restrict__ W2,
                              unsigned short* __restrict__ Wt0,
                              unsigned short* __restrict__ Wt1,
                              unsigned short* __restrict__ Wt2) {
    int e = blockIdx.x * 256 + threadIdx.x;
    const int s0 = 1024 * 512;
    const int s1 = s0 + 1024 * 768;
    if (e < s0) {
        int j = e / 512, k = e - j * 512;
        Wt0[e] = f2bf(W0[k * 1024 + j]);
    } else if (e < s1) {
        int i2 = e - s0;
        int j = i2 / 768, k = i2 - j * 768;
        Wt1[i2] = f2bf(W1[k * 1024 + j]);
    } else {
        int i2 = e - s1;
        int j = i2 / 768, k = i2 - j * 768;
        Wt2[i2] = f2bf(W2[k * 1024 + j]);
    }
}

// ---------------------------------------------------------------------------
// K1: embedding gather + fp32->bf16.  Row r = t*32+b.  Grid 64000 x 256.
// ---------------------------------------------------------------------------
__global__ void gather_x_kernel(const int* __restrict__ inputs,
                                const float* __restrict__ emb,
                                unsigned short* __restrict__ Xbf) {
    int e = blockIdx.x * 256 + threadIdx.x;
    int r = e >> 8;
    int k = e & 255;
    int b = r & 31;
    int t = r >> 5;
    int vid = inputs[b * TSTEPS + t];
    Xbf[e] = f2bf(emb[(size_t)vid * SZ + k]);
}

// ---------------------------------------------------------------------------
// K2: phase-1 GEMM  Zx_l = X @ W_l[0:256,:] + b_l  (bf16 out, fp32 accum)
// ---------------------------------------------------------------------------
__global__ __launch_bounds__(256) void phase1_kernel(
    const unsigned short* __restrict__ Xbf,
    const unsigned short* __restrict__ Wt0,
    const unsigned short* __restrict__ Wt1,
    const unsigned short* __restrict__ Wt2,
    const float* __restrict__ b0,
    const float* __restrict__ b1,
    const float* __restrict__ b2,
    unsigned short* __restrict__ Zx) {
    int lid   = blockIdx.x;
    int layer = lid / 8000;
    int rem   = lid - layer * 8000;
    int mt    = rem >> 1;
    int nh    = rem & 1;

    const unsigned short* Wt = (layer == 0) ? Wt0 : (layer == 1) ? Wt1 : Wt2;
    const float* bias        = (layer == 0) ? b0  : (layer == 1) ? b1  : b2;
    const int ldK            = (layer == 0) ? 512 : 768;

    int wave = threadIdx.x >> 5;
    int lane = threadIdx.x & 31;
    int r0    = mt * 16;
    int cbase = nh * 512 + wave * 64;

    int arow = lane & 15, ahalf = lane >> 4;
    int bn   = lane & 15, bh    = lane >> 4;

    v8f acc[4] = {};
    const unsigned short* Ap = Xbf + (size_t)(r0 + arow) * SZ + ahalf * 8;
    for (int k = 0; k < 256; k += 32) {
        BF16Frag a;
        a.q[0] = *(const int4*)(Ap + k);
        a.q[1] = *(const int4*)(Ap + k + 16);
#pragma unroll
        for (int nt = 0; nt < 4; ++nt) {
            const unsigned short* Bp =
                Wt + (size_t)(cbase + nt * 16 + bn) * ldK + k + bh * 8;
            BF16Frag b;
            b.q[0] = *(const int4*)(Bp);
            b.q[1] = *(const int4*)(Bp + 16);
            acc[nt] = wmma_bf16(a.v, b.v, acc[nt]);
        }
    }

    size_t zbase = (size_t)layer * ROWS * NG;
#pragma unroll
    for (int nt = 0; nt < 4; ++nt) {
        int col  = cbase + nt * 16 + (lane & 15);
        float bv = bias[col];
#pragma unroll
        for (int v = 0; v < 8; ++v) {
            int row = r0 + v + (lane >> 4) * 8;
            Zx[zbase + (size_t)row * NG + col] = f2bf(acc[nt][v] + bv);
        }
    }
}

// ---------------------------------------------------------------------------
// Recurrent (32,256)@(256,1024) GEMM block accumulated into acc[2][2].
// ---------------------------------------------------------------------------
__device__ __forceinline__ void gemm_k256(
    const unsigned short* __restrict__ hbase,   // LDS [32][256] bf16
    const unsigned short* __restrict__ Wt,      // global [1024][ldK] bf16
    int ldK, int krow0, int nbase,
    int arow, int ahalf, int bn, int bh,
    v8f acc[2][2]) {
    const unsigned short* a0p = hbase + arow * SZ + ahalf * 8;
    const unsigned short* a1p = hbase + (16 + arow) * SZ + ahalf * 8;
    const unsigned short* b0p = Wt + (size_t)(nbase + bn) * ldK + krow0 + bh * 8;
    const unsigned short* b1p = Wt + (size_t)(nbase + 16 + bn) * ldK + krow0 + bh * 8;
#pragma unroll
    for (int k = 0; k < 256; k += 32) {
        BF16Frag a0, a1, b0, b1;
        a0.q[0] = *(const int4*)(a0p + k);
        a0.q[1] = *(const int4*)(a0p + k + 16);
        a1.q[0] = *(const int4*)(a1p + k);
        a1.q[1] = *(const int4*)(a1p + k + 16);
        b0.q[0] = *(const int4*)(b0p + k);
        b0.q[1] = *(const int4*)(b0p + k + 16);
        b1.q[0] = *(const int4*)(b1p + k);
        b1.q[1] = *(const int4*)(b1p + k + 16);
        acc[0][0] = wmma_bf16(a0.v, b0.v, acc[0][0]);
        acc[0][1] = wmma_bf16(a0.v, b1.v, acc[0][1]);
        acc[1][0] = wmma_bf16(a1.v, b0.v, acc[1][0]);
        acc[1][1] = wmma_bf16(a1.v, b1.v, acc[1][1]);
    }
}

// ---------------------------------------------------------------------------
// K3: sequential recurrence, one full-WGP workgroup (32 wave32s).
// Dynamic LDS layout (bytes):
//   [0,      65536)  zbuf0  (TDM stage, even stages)
//   [65536, 131072)  zbuf1  (TDM stage, odd stages)
//   [131072,262144)  z_sh   (f32 cross-wave gate exchange)
// ---------------------------------------------------------------------------
__global__ __launch_bounds__(1024) void lstm_seq_kernel(
    const unsigned short* __restrict__ Zx,     // [3][64000][1024] bf16
    const unsigned short* __restrict__ Wt0,
    const unsigned short* __restrict__ Wt1,
    const unsigned short* __restrict__ Wt2,
    const int* __restrict__ nstarts,
    float* __restrict__ out,                   // [32][768]
    float* __restrict__ c_glob)                // [3][32][256]
{
    __shared__ unsigned short h_sh[3 * BATCHN * SZ];          // 48KB static
    extern __shared__ unsigned short zstage[];                // 256KB dynamic
    const unsigned LDS_DYN_BASE = 3 * BATCHN * SZ * 2;        // 48KB
    unsigned short* zbuf0 = zstage;
    unsigned short* zbuf1 = zstage + BATCHN * NG;             // +32768 elems
    float* z_sh = (float*)(zstage + 2 * BATCHN * NG);         // +128KB bytes

    int tid = threadIdx.x;
    for (int i = tid; i < 3 * BATCHN * SZ; i += 1024) {
        h_sh[i]   = 0;
        c_glob[i] = 0.0f;
    }
    __syncthreads();

    int wave = tid >> 5, lane = tid & 31;
    int nbase = wave * 32;
    int arow = lane & 15, ahalf = lane >> 4;
    int bn   = lane & 15, bh    = lane >> 4;

    int eb  = tid >> 5;
    int eu0 = lane * 8;
    int tb  = nstarts[eb * 3 + 0];

#if HAVE_TDM
    if (wave == 0) tdm_load_tile(Zx, LDS_DYN_BASE);   // stage 0 -> zbuf0
#endif

    for (int t = 0; t < TSTEPS; ++t) {
        for (int l = 0; l < 3; ++l) {
            int s = t * 3 + l;
            size_t cur = ((size_t)l * ROWS + (size_t)t * BATCHN) * NG;

#if HAVE_TDM
            // Kick the DMA for the NEXT stage; overlaps the WMMA work below.
            if (wave == 0 && (s + 1 < NSTAGE)) {
                size_t nxt = (l < 2) ? (cur + (size_t)ROWS * NG)
                                     : ((size_t)(t + 1) * BATCHN * NG);
                tdm_load_tile(Zx + nxt,
                              LDS_DYN_BASE + (((s + 1) & 1) ? 65536u : 0u));
            }
#endif
            v8f acc[2][2] = {};
            if (l == 0) {
                gemm_k256(&h_sh[0], Wt0, 512, 256, nbase,
                          arow, ahalf, bn, bh, acc);
            } else if (l == 1) {
                gemm_k256(&h_sh[0],               Wt1, 768, 256, nbase,
                          arow, ahalf, bn, bh, acc);
                gemm_k256(&h_sh[BATCHN * SZ],     Wt1, 768, 512, nbase,
                          arow, ahalf, bn, bh, acc);
            } else {
                gemm_k256(&h_sh[BATCHN * SZ],     Wt2, 768, 256, nbase,
                          arow, ahalf, bn, bh, acc);
                gemm_k256(&h_sh[2 * BATCHN * SZ], Wt2, 768, 512, nbase,
                          arow, ahalf, bn, bh, acc);
            }

            // scatter recurrent partial z into the LDS exchange buffer
#pragma unroll
            for (int mt = 0; mt < 2; ++mt)
#pragma unroll
                for (int nt = 0; nt < 2; ++nt)
#pragma unroll
                    for (int v = 0; v < 8; ++v) {
                        int m = mt * 16 + v + (lane >> 4) * 8;
                        int n = nbase + nt * 16 + (lane & 15);
                        z_sh[m * NG + n] = acc[mt][nt][v];
                    }

#if HAVE_TDM
            // Retire the current stage's DMA (next stage's may stay in flight:
            // TDM ops are in-order per wave, so tensorcnt<=1 == current done).
            if (wave == 0) {
                if (s + 1 < NSTAGE) __builtin_amdgcn_s_wait_tensorcnt(1);
                else                __builtin_amdgcn_s_wait_tensorcnt(0);
            }
#endif
            __threadfence_block();
            __syncthreads();

#if HAVE_TDM
            const unsigned short* Zrow =
                ((s & 1) ? zbuf1 : zbuf0) + eb * NG;
#else
            const unsigned short* Zrow = Zx + cur + (size_t)eb * NG;
#endif
            float* crow = c_glob + (l * BATCHN + eb) * SZ;
            unsigned short* hrow = h_sh + l * BATCHN * SZ + eb * SZ;
            const float* zrow = z_sh + eb * NG;

#pragma unroll
            for (int uu = 0; uu < 8; ++uu) {
                int u = eu0 + uu;
                float zi = zrow[u]       + bf2f(Zrow[u]);
                float zj = zrow[256 + u] + bf2f(Zrow[256 + u]);
                float zf = zrow[512 + u] + bf2f(Zrow[512 + u]);
                float zo = zrow[768 + u] + bf2f(Zrow[768 + u]);
                float c  = crow[u];
                float cn = c * sigm(zf) + sigm(zi) * ftanh(zj);
                float hn = ftanh(cn) * sigm(zo);
                crow[u] = cn;
                hrow[u] = f2bf(hn);
                if (t == tb) {
                    if (l == 0) {
                        out[eb * 768 + u]       = cn;   // c0
                        out[eb * 768 + 256 + u] = hn;   // h0
                    } else if (l == 1) {
                        out[eb * 768 + 512 + u] = cn;   // c1
                    }
                }
            }
            __threadfence_block();
            __syncthreads();
        }
    }
}

// ---------------------------------------------------------------------------
extern "C" void kernel_launch(void* const* d_in, const int* in_sizes, int n_in,
                              void* d_out, int out_size, void* d_ws, size_t ws_size,
                              hipStream_t stream) {
    (void)in_sizes; (void)n_in; (void)out_size;

    const int*   inputs  = (const int*)  d_in[0];
    const int*   nstarts = (const int*)  d_in[1];
    const float* emb     = (const float*)d_in[2];
    const float* W0      = (const float*)d_in[3];
    const float* b0      = (const float*)d_in[4];
    const float* W1      = (const float*)d_in[5];
    const float* b1      = (const float*)d_in[6];
    const float* W2      = (const float*)d_in[7];
    const float* b2      = (const float*)d_in[8];

    char* ws = (char*)d_ws;
    size_t oWt0 = 0;
    size_t oWt1 = oWt0 + (size_t)1024 * 512 * 2;
    size_t oWt2 = oWt1 + (size_t)1024 * 768 * 2;
    size_t oX   = oWt2 + (size_t)1024 * 768 * 2;
    size_t oZx  = oX   + (size_t)ROWS * SZ * 2;
    size_t oC   = oZx  + (size_t)3 * ROWS * NG * 2;
    size_t need = oC   + (size_t)3 * BATCHN * SZ * 4;
    if (ws_size < need) return;   // ~430 MB workspace required

    unsigned short* Wt0 = (unsigned short*)(ws + oWt0);
    unsigned short* Wt1 = (unsigned short*)(ws + oWt1);
    unsigned short* Wt2 = (unsigned short*)(ws + oWt2);
    unsigned short* Xbf = (unsigned short*)(ws + oX);
    unsigned short* Zx  = (unsigned short*)(ws + oZx);
    float* c_glob = (float*)(ws + oC);

    const int dynLds = (2 * BATCHN * NG * 2) + (BATCHN * NG * 4);  // 256 KB
    (void)hipFuncSetAttribute((const void*)lstm_seq_kernel,
                              hipFuncAttributeMaxDynamicSharedMemorySize,
                              dynLds);

    conv_w_kernel  <<<8192,  256, 0, stream>>>(W0, W1, W2, Wt0, Wt1, Wt2);
    gather_x_kernel<<<ROWS,  256, 0, stream>>>(inputs, emb, Xbf);
    phase1_kernel  <<<24000, 256, 0, stream>>>(Xbf, Wt0, Wt1, Wt2,
                                               b0, b1, b2, Zx);
    lstm_seq_kernel<<<1, 1024, dynLds, stream>>>(Zx, Wt0, Wt1, Wt2, nstarts,
                                                 (float*)d_out, c_glob);
}
